// WindowAttention_19782619365403
// MI455X (gfx1250) — matrix operand order repaired and checked by reading, not compiled, not run on hardware
//
#include <hip/hip_runtime.h>
#include <hip/hip_bf16.h>

typedef __attribute__((ext_vector_type(16))) __bf16 v16bf;
typedef __attribute__((ext_vector_type(8)))  float  v8f;

#define NWIN   2048
#define NTOK   49
#define CDIM   384
#define NHEAD  12
#define HDIM   32
#define ROWS   100352          /* 2048*49 == 1568*64 exactly */
#define ROWS_P 100416          /* +64 pad rows for 49->64 tile over-read */
#define QKVC   1152
#define SCALEQ 0.17677669529663689f   /* 32^-0.5 */

union BF16x16 { v16bf v; uint4 q[2]; __bf16 e[16]; };

__device__ inline v8f vzero8() {
    v8f z;
#pragma unroll
    for (int i = 0; i < 8; ++i) z[i] = 0.f;
    return z;
}

// A-matrix 16x32 bf16 fragment: row = lane&15,
// 16 contiguous K values at [2*off..2*off+15], off = (lane>=16)?8:0
__device__ inline v16bf fragA_bf16(const __bf16* rowp, int off) {
    BF16x16 u;
    u.q[0] = *(const uint4*)(rowp + 2 * off);
    u.q[1] = *(const uint4*)(rowp + 2 * off + 8);
    return u.v;
}
// B-matrix 32x16 bf16 fragment: col = lane&15, 16 contiguous K values
// starting at (lane>=16 ? 16 : 0) -- caller bakes that into p.
__device__ inline v16bf fragB_bf16(const __bf16* p) {
    BF16x16 u;
    u.q[0] = *(const uint4*)(p);
    u.q[1] = *(const uint4*)(p + 8);
    return u.v;
}

__device__ inline v8f wmma_bf16(v16bf a, v16bf b, v8f c) {
    return __builtin_amdgcn_wmma_f32_16x16x32_bf16(false, a, false, b, (short)0, c, false, false);
}

// ---- CDNA5 async global->LDS copy (ISA 15.18.3 op 98, tracked by ASYNCcnt) --
// LDS destination address = low 32 bits of the generic shared pointer
// (ISA 10.2: flat LDS address[31:0] is the wave-relative LDS offset).
__device__ inline void async_copy16(const __bf16* g, __bf16* l) {
    unsigned lo = (unsigned)(size_t)(void*)l;
    asm volatile("global_load_async_to_lds_b128 %0, %1, off"
                 :: "v"(lo), "v"((unsigned long long)(size_t)g)
                 : "memory");
}
__device__ inline void wait_async0() {
    asm volatile("s_wait_asynccnt 0" ::: "memory");
}

__device__ inline float rmax16(float v) {
#pragma unroll
    for (int o = 1; o < 16; o <<= 1) v = fmaxf(v, __shfl_xor(v, o, 32));
    return v;
}
__device__ inline float rsum16(float v) {
#pragma unroll
    for (int o = 1; o < 16; o <<= 1) v += __shfl_xor(v, o, 32);
    return v;
}

// ---------------- Kernel 0a: fp32 -> bf16 bulk convert (n multiple of 8) ----
__global__ __launch_bounds__(256)
void cvt_bf16_kernel(const float* __restrict__ src, __bf16* __restrict__ dst,
                     long n) {
    long i = ((long)blockIdx.x * 256 + threadIdx.x) * 8;
    if (i >= n) return;
    float4 f0 = *(const float4*)(src + i);
    float4 f1 = *(const float4*)(src + i + 4);
    BF16x16 u;
    u.e[0]=(__bf16)f0.x; u.e[1]=(__bf16)f0.y; u.e[2]=(__bf16)f0.z; u.e[3]=(__bf16)f0.w;
    u.e[4]=(__bf16)f1.x; u.e[5]=(__bf16)f1.y; u.e[6]=(__bf16)f1.z; u.e[7]=(__bf16)f1.w;
    *(uint4*)(dst + i) = u.q[0];
}

// ---------------- Kernel 0b: bias+mask table, padded to 64x64 ---------------
__global__ __launch_bounds__(256)
void bm_kernel(const float* __restrict__ relt, const float* __restrict__ mask,
               float* __restrict__ bm) {
    int h = blockIdx.x / 64, w = blockIdx.x % 64;
    float* o = bm + ((size_t)(h * 64 + w)) * 4096;
    for (int idx = threadIdx.x; idx < 4096; idx += 256) {
        int n = idx >> 6, m = idx & 63;
        float v = -1e30f;
        if (n < NTOK && m < NTOK) {
            int ih = n / 7, iw = n % 7, jh = m / 7, jw = m % 7;
            int rel = (ih - jh + 6) * 13 + (iw - jw + 6);
            v = relt[rel * NHEAD + h] + mask[(w * NTOK + n) * NTOK + m];
        }
        o[idx] = v;
    }
}

// ---------------- Kernel 1: QKV GEMM (bf16 in, async double-buffered A) -----
__global__ __launch_bounds__(128)
void qkv_gemm(const __bf16* __restrict__ xb, const __bf16* __restrict__ w,
              const float* __restrict__ bias, __bf16* __restrict__ qkv) {
    __shared__ __align__(16) __bf16 As[2][64 * 40];
    const int tid = threadIdx.x, wave = tid >> 5, lane = tid & 31;
    const int ln = lane & 15, hi = lane >> 4;
    const int offA = hi * 8, dlo = hi * 16;
    const int row0 = blockIdx.x * 64;
    const int j = blockIdx.y * 64 + wave * 16 + ln;   // output col = weight row
    const __bf16* wrow = w + (size_t)j * CDIM;
    const int sr = tid >> 1, sc = (tid & 1) * 16;
    const __bf16* xrow = xb + (size_t)(row0 + sr) * CDIM + sc;
    __bf16* lrow0 = &As[0][sr * 40 + sc];
    __bf16* lrow1 = &As[1][sr * 40 + sc];

    // prologue: async-stage tile 0
    async_copy16(xrow, lrow0);
    async_copy16(xrow + 8, lrow0 + 8);

    v8f acc[4];
#pragma unroll
    for (int m = 0; m < 4; ++m) acc[m] = vzero8();

#pragma unroll
    for (int ks = 0; ks < 12; ++ks) {
        wait_async0();        // own tile-ks copies landed in LDS
        __syncthreads();      // everyone's landed; prev buffer reads all done
        if (ks + 1 < 12) {    // overlap next tile's async copy with WMMAs
            const __bf16* nx = xrow + (ks + 1) * 32;
            __bf16* nl = (ks & 1) ? lrow0 : lrow1;
            async_copy16(nx, nl);
            async_copy16(nx + 8, nl + 8);
        }
        v16bf bf = fragB_bf16(wrow + ks * 32 + dlo);
        const __bf16* cbuf = As[ks & 1];
#pragma unroll
        for (int m = 0; m < 4; ++m) {
            v16bf af = fragA_bf16(&cbuf[(m * 16 + ln) * 40], offA);
            acc[m] = wmma_bf16(af, bf, acc[m]);
        }
    }
    const float bj = bias[j];
    const float sc2 = (j < CDIM) ? SCALEQ : 1.0f;   // pre-scale q by D^-0.5
#pragma unroll
    for (int m = 0; m < 4; ++m)
#pragma unroll
        for (int g = 0; g < 8; ++g) {
            int r = row0 + m * 16 + g + hi * 8;
            qkv[(size_t)r * QKVC + j] = (__bf16)((acc[m][g] + bj) * sc2);
        }
}

// ---------------- Kernel 2: per-(window,head) attention ---------------------
__global__ __launch_bounds__(32)
void attn_kernel(const __bf16* __restrict__ qkv, const float* __restrict__ bm,
                 __bf16* __restrict__ obuf) {
    const int b = blockIdx.x / NHEAD, h = blockIdx.x % NHEAD, wwin = b & 63;
    const int lane = threadIdx.x & 31, ln = lane & 15, hi = lane >> 4;
    const int offA = hi * 8, dlo = hi * 16;
    __shared__ __align__(16) __bf16 vlds[64 * 40];
    __shared__ __align__(16) __bf16 Pl[16 * 72];

    const __bf16* base = qkv + (size_t)b * NTOK * QKVC + h * HDIM;

    // async-stage V (49 rows x 32 bf16 = 196 x 16B chunks) straight into LDS
    for (int t = lane; t < 196; t += 32) {
        int r = t >> 2, c = (t & 3) * 8;
        async_copy16(base + (size_t)r * QKVC + 2 * CDIM + c, &vlds[r * 40 + c]);
    }
    // zero-pad rows 49..63 (their probs are exactly 0, keep operands finite)
    for (int r = 49 + lane; r < 64; r += 32) {
        uint4 z = {0, 0, 0, 0};
        uint4* d = (uint4*)&vlds[r * 40];
        d[0] = z; d[1] = z; d[2] = z; d[3] = z;
    }
    wait_async0();

    // preload the 4 V B-fragments (K=m tiles x N=d tiles), reused by all mtiles
    v16bf vf[2][2];
#pragma unroll
    for (int ks = 0; ks < 2; ++ks)
#pragma unroll
        for (int nt = 0; nt < 2; ++nt) {
            BF16x16 u;
            int d = nt * 16 + ln, mb = ks * 32 + dlo;
#pragma unroll
            for (int i = 0; i < 16; ++i) u.e[i] = vlds[(mb + i) * 40 + d];
            vf[ks][nt] = u.v;
        }

    const float* bmb = bm + ((size_t)(h * 64 + wwin)) * 4096;

#pragma unroll
    for (int mt = 0; mt < 4; ++mt) {
        // scores = q @ k^T (q pre-scaled); single K=32 WMMA step
        v8f acc[4];
#pragma unroll
        for (int nt = 0; nt < 4; ++nt) acc[nt] = vzero8();
        v16bf aq = fragA_bf16(base + (size_t)(mt * 16 + ln) * QKVC, offA);
#pragma unroll
        for (int nt = 0; nt < 4; ++nt) {
            v16bf bk = fragB_bf16(base + (size_t)(nt * 16 + ln) * QKVC + CDIM + dlo);
            acc[nt] = wmma_bf16(aq, bk, acc[nt]);
        }
        // + rel-bias + mask (padded with -1e30 for m>=49)
#pragma unroll
        for (int nt = 0; nt < 4; ++nt)
#pragma unroll
            for (int g = 0; g < 8; ++g)
                acc[nt][g] += bmb[(mt * 16 + g + hi * 8) * 64 + nt * 16 + ln];
        // row softmax: each row lives in 16 lanes of one half-wave
#pragma unroll
        for (int g = 0; g < 8; ++g) {
            float mx = fmaxf(fmaxf(acc[0][g], acc[1][g]), fmaxf(acc[2][g], acc[3][g]));
            mx = rmax16(mx);
            float s = 0.f;
#pragma unroll
            for (int nt = 0; nt < 4; ++nt) {
                float p = __expf(acc[nt][g] - mx);
                acc[nt][g] = p; s += p;
            }
            s = rsum16(s);
            float inv = 1.0f / s;
#pragma unroll
            for (int nt = 0; nt < 4; ++nt)
                Pl[(g + hi * 8) * 72 + nt * 16 + ln] = (__bf16)(acc[nt][g] * inv);
        }
        // out = P(16x64) @ V(64x32)
        v8f o[2];
        o[0] = vzero8(); o[1] = vzero8();
#pragma unroll
        for (int ks = 0; ks < 2; ++ks) {
            v16bf ap = fragA_bf16(&Pl[ln * 72 + ks * 32], offA);
            o[0] = wmma_bf16(ap, vf[ks][0], o[0]);
            o[1] = wmma_bf16(ap, vf[ks][1], o[1]);
        }
#pragma unroll
        for (int nt = 0; nt < 2; ++nt)
#pragma unroll
            for (int g = 0; g < 8; ++g) {
                int n = mt * 16 + g + hi * 8;
                if (n < NTOK)
                    obuf[(size_t)(b * NTOK + n) * CDIM + h * HDIM + nt * 16 + ln] =
                        (__bf16)o[nt][g];
            }
    }
}

// ---------------- Kernel 3: output projection (bf16 in, fp32 out) ----------
__global__ __launch_bounds__(128)
void proj_gemm(const __bf16* __restrict__ a, const __bf16* __restrict__ w,
               const float* __restrict__ bias, float* __restrict__ out) {
    __shared__ __align__(16) __bf16 As[2][64 * 40];
    const int tid = threadIdx.x, wave = tid >> 5, lane = tid & 31;
    const int ln = lane & 15, hi = lane >> 4;
    const int offA = hi * 8, dlo = hi * 16;
    const int row0 = blockIdx.x * 64;
    const int j = blockIdx.y * 64 + wave * 16 + ln;
    const __bf16* wrow = w + (size_t)j * CDIM;
    const int sr = tid >> 1, sc = (tid & 1) * 16;
    const __bf16* arow = a + (size_t)(row0 + sr) * CDIM + sc;
    __bf16* lrow0 = &As[0][sr * 40 + sc];
    __bf16* lrow1 = &As[1][sr * 40 + sc];

    async_copy16(arow, lrow0);
    async_copy16(arow + 8, lrow0 + 8);

    v8f acc[4];
#pragma unroll
    for (int m = 0; m < 4; ++m) acc[m] = vzero8();

#pragma unroll
    for (int ks = 0; ks < 12; ++ks) {
        wait_async0();
        __syncthreads();
        if (ks + 1 < 12) {
            const __bf16* nx = arow + (ks + 1) * 32;
            __bf16* nl = (ks & 1) ? lrow0 : lrow1;
            async_copy16(nx, nl);
            async_copy16(nx + 8, nl + 8);
        }
        v16bf bf = fragB_bf16(wrow + ks * 32 + dlo);
        const __bf16* cbuf = As[ks & 1];
#pragma unroll
        for (int m = 0; m < 4; ++m) {
            v16bf af = fragA_bf16(&cbuf[(m * 16 + ln) * 40], offA);
            acc[m] = wmma_bf16(af, bf, acc[m]);
        }
    }
    const float bj = bias[j];
#pragma unroll
    for (int m = 0; m < 4; ++m)
#pragma unroll
        for (int g = 0; g < 8; ++g) {
            int r = row0 + m * 16 + g + hi * 8;
            out[(size_t)r * CDIM + j] = acc[m][g] + bj;
        }
}

extern "C" void kernel_launch(void* const* d_in, const int* in_sizes, int n_in,
                              void* d_out, int out_size, void* d_ws, size_t ws_size,
                              hipStream_t stream) {
    const float* x      = (const float*)d_in[0];
    const float* mask   = (const float*)d_in[1];
    const float* qkv_w  = (const float*)d_in[2];
    const float* qkv_b  = (const float*)d_in[3];
    const float* proj_w = (const float*)d_in[4];
    const float* proj_b = (const float*)d_in[5];
    const float* relt   = (const float*)d_in[6];
    float* out = (float*)d_out;

    char* ws = (char*)d_ws;
    const size_t XBF_BYTES  = (size_t)ROWS * CDIM * sizeof(__bf16);        // 77 MB
    const size_t WQ_BYTES   = (size_t)QKVC * CDIM * sizeof(__bf16);        // 884 KB
    const size_t WP_BYTES   = (size_t)CDIM * CDIM * sizeof(__bf16);        // 295 KB
    const size_t QKV_BYTES  = (size_t)ROWS_P * QKVC * sizeof(__bf16);      // 231 MB
    const size_t BM_BYTES   = (size_t)NHEAD * 64 * 4096 * sizeof(float);   // 12.6 MB
    __bf16* xbf  = (__bf16*)ws;
    __bf16* wqb  = (__bf16*)(ws + XBF_BYTES);
    __bf16* wpb  = (__bf16*)(ws + XBF_BYTES + WQ_BYTES);
    __bf16* qkv  = (__bf16*)(ws + XBF_BYTES + WQ_BYTES + WP_BYTES);
    float*  bm   = (float*)(ws + XBF_BYTES + WQ_BYTES + WP_BYTES + QKV_BYTES);
    __bf16* obuf = (__bf16*)(ws + XBF_BYTES + WQ_BYTES + WP_BYTES + QKV_BYTES + BM_BYTES);

    const long nx = (long)ROWS * CDIM;        // 38,535,168 (mult of 8)
    const long nwq = (long)QKVC * CDIM;       // 442,368
    const long nwp = (long)CDIM * CDIM;       // 147,456
    cvt_bf16_kernel<<<(unsigned)(nx / 8 / 256), 256, 0, stream>>>(x, xbf, nx);
    cvt_bf16_kernel<<<(unsigned)(nwq / 8 / 256), 256, 0, stream>>>(qkv_w, wqb, nwq);
    cvt_bf16_kernel<<<(unsigned)(nwp / 8 / 256), 256, 0, stream>>>(proj_w, wpb, nwp);
    bm_kernel <<<NHEAD * 64, 256, 0, stream>>>(relt, mask, bm);

    qkv_gemm  <<<dim3(ROWS / 64, QKVC / 64), 128, 0, stream>>>(xbf, wqb, qkv_b, qkv);
    attn_kernel<<<NWIN * NHEAD, 32, 0, stream>>>(qkv, bm, obuf);
    proj_gemm <<<dim3(ROWS / 64, CDIM / 64), 128, 0, stream>>>(obuf, wpb, proj_b, out);
}
